// GSSelfAttn_20942260535581
// MI455X (gfx1250) — matrix operand order repaired and checked by reading, not compiled, not run on hardware
//
#include <hip/hip_runtime.h>
#include <hip/hip_bf16.h>

typedef __bf16 bf16;
typedef __attribute__((ext_vector_type(16))) __bf16 v16bf;
typedef __attribute__((ext_vector_type(8)))  float  v8f;

// ---------------------------------------------------------------------------
// Geometry
//   DIM=180, HEADS=6, HD=30 (padded to 32), N=144 (= 9 tiles of 16), B=2048
//   Padded dims: DP=192 (K for projections), NPK=160 (K for P@V)
// ---------------------------------------------------------------------------
#define NWIN   2048
#define NTOK   144
#define DIM_   180
#define DP     192
#define HEADS_ 6
#define HD_    30
#define NPK    160

// LDS layout (bytes)
#define OFF_GSB   0            // bf16 [144][192]  = 55296
#define OFF_QH    55296        // bf16 [144][32]   =  9216
#define OFF_KH    64512        // bf16 [144][32]   =  9216
#define OFF_VHT   73728        // bf16 [32][160]   = 10240
#define OFF_PBUF  83968        // bf16 [144][160]  = 46080
#define OFF_OB    130048       // bf16 [144][192]  = 55296
#define SMEM_BYTES 185344

// workspace layout (bytes)
#define WS_WQT 0
#define WS_WKT 73728
#define WS_WVT 147456
#define WS_WPT 221184
#define WS_RELB 294912

static __device__ __forceinline__ v16bf ld16(const bf16* p0, const bf16* p1) {
    union { v16bf v; uint4 q[2]; } u;
    u.q[0] = *(const uint4*)p0;
    u.q[1] = *(const uint4*)p1;
    return u.v;
}

static __device__ __forceinline__ v8f wmma_bf16(v16bf a, v16bf b, v8f c) {
    return __builtin_amdgcn_wmma_f32_16x16x32_bf16(
        /*neg_a=*/false, a, /*neg_b=*/false, b,
        /*c_mod=*/(short)0, c, /*reuse_a=*/false, /*reuse_b=*/false);
}

// xor-reductions within each 16-lane half (C/D fragment rows never cross the
// lane16 boundary, so masks 1,2,4,8 are exactly right).
static __device__ __forceinline__ float red_max16(float x) {
    x = fmaxf(x, __int_as_float(__builtin_amdgcn_ds_swizzle(__float_as_int(x), 0x041F)));
    x = fmaxf(x, __int_as_float(__builtin_amdgcn_ds_swizzle(__float_as_int(x), 0x081F)));
    x = fmaxf(x, __int_as_float(__builtin_amdgcn_ds_swizzle(__float_as_int(x), 0x101F)));
    x = fmaxf(x, __int_as_float(__builtin_amdgcn_ds_swizzle(__float_as_int(x), 0x201F)));
    return x;
}
static __device__ __forceinline__ float red_add16(float x) {
    x += __int_as_float(__builtin_amdgcn_ds_swizzle(__float_as_int(x), 0x041F));
    x += __int_as_float(__builtin_amdgcn_ds_swizzle(__float_as_int(x), 0x081F));
    x += __int_as_float(__builtin_amdgcn_ds_swizzle(__float_as_int(x), 0x101F));
    x += __int_as_float(__builtin_amdgcn_ds_swizzle(__float_as_int(x), 0x201F));
    return x;
}

// ---------------------------------------------------------------------------
// Compile-time-specialized projection pass: MAT 0=Q (scaled), 1=K, 2=V^T.
// Each wave computes 2 of the 18 [16x16] output tiles (t = wave, wave+9).
// No data-dependent branches -> no EXEC juggling around the LDS stores.
// ---------------------------------------------------------------------------
template <int MAT>
static __device__ __forceinline__ void proj_pass(
    int wave, int l16, int hi, int h,
    const bf16* __restrict__ gsb, const bf16* __restrict__ wT,
    const float* __restrict__ bias, bf16* __restrict__ dst) {
    #pragma unroll
    for (int i = 0; i < 2; ++i) {
        int t = wave + 9 * i;              // 0..17
        int mt = t >> 1, nt = t & 1;
        int nloc = nt * 16 + l16;          // 0..31 within padded head
        int ncol = h * HD_ + nloc;         // < 192 always
        float bval = (nloc < HD_) ? bias[ncol] : 0.0f;
        v8f acc;
        #pragma unroll
        for (int r = 0; r < 8; ++r) acc[r] = bval;
        const bf16* wrow = wT  + ncol * DP;
        const bf16* arow = gsb + (mt * 16 + l16) * DP;
        #pragma unroll
        for (int kk = 0; kk < 6; ++kk) {
            v16bf a  = ld16(arow + kk * 32 + hi * 8,  arow + kk * 32 + hi * 8 + 16);
            v16bf bm = ld16(wrow + kk * 32 + hi * 8,  wrow + kk * 32 + hi * 8 + 16);
            acc = wmma_bf16(a, bm, acc);
        }
        int mbase = mt * 16 + hi * 8;
        if (MAT == 0) {                     // q *= hd^-0.5 (bias included)
            #pragma unroll
            for (int r = 0; r < 8; ++r) acc[r] *= 0.18257418583505537f;
        }
        if (MAT == 2) {
            // V^T: dst = vht[32][160]; consecutive r -> consecutive addresses,
            // pack two bf16 per 32-bit store.
            #pragma unroll
            for (int r = 0; r < 8; r += 2) {
                union { unsigned u; bf16 hh[2]; } pk;
                pk.hh[0] = (bf16)acc[r];
                pk.hh[1] = (bf16)acc[r + 1];
                *(unsigned*)(dst + nloc * NPK + mbase + r) = pk.u;
            }
        } else {
            // Q/K: row-major [144][32], column store (stride 32).
            #pragma unroll
            for (int r = 0; r < 8; ++r)
                dst[(mbase + r) * 32 + nloc] = (bf16)acc[r];
        }
    }
}

// ---------------------------------------------------------------------------
// Prep: transpose + bf16-convert + zero-pad the 4 weight matrices to [192][192]
// (row n = output column, contiguous K), and expand bias_table[REL_IDX] into a
// dense [6][144][144] f32 table.
// ---------------------------------------------------------------------------
__global__ void gs_attn_prep(const float* __restrict__ wq, const float* __restrict__ wk,
                             const float* __restrict__ wv, const float* __restrict__ wp,
                             const float* __restrict__ bt,
                             bf16* __restrict__ wqT, bf16* __restrict__ wkT,
                             bf16* __restrict__ wvT, bf16* __restrict__ wpT,
                             float* __restrict__ relb) {
    const int TW = DP * DP;
    const int total1 = 4 * TW;
    const int total  = total1 + HEADS_ * NTOK * NTOK;
    int idx = blockIdx.x * 256 + threadIdx.x;
    if (idx >= total) return;
    if (idx < total1) {
        int w = idx / TW, rem = idx % TW;
        int n = rem / DP, k = rem % DP;
        const float* W = (w == 0) ? wq : (w == 1) ? wk : (w == 2) ? wv : wp;
        bf16* D = (w == 0) ? wqT : (w == 1) ? wkT : (w == 2) ? wvT : wpT;
        float v = (n < DIM_ && k < DIM_) ? W[k * DIM_ + n] : 0.0f;
        D[n * DP + k] = (bf16)v;
    } else {
        int r = idx - total1;
        int h = r / (NTOK * NTOK), qk = r % (NTOK * NTOK);
        int q = qk / NTOK, k2 = qk % NTOK;
        int qy = q / 12, qx = q % 12, ky = k2 / 12, kx = k2 % 12;
        int ridx = (qy - ky + 11) * 23 + (qx - kx + 11);
        relb[h * NTOK * NTOK + q * NTOK + k2] = bt[ridx * HEADS_ + h];
    }
}

// ---------------------------------------------------------------------------
// Fused attention: one workgroup (288 thr = 9 wave32) per window.
// ---------------------------------------------------------------------------
__global__ __launch_bounds__(288) void gs_attn_fused(
    const float* __restrict__ gs,
    const float* __restrict__ bq, const float* __restrict__ bk,
    const float* __restrict__ bv, const float* __restrict__ bp,
    const bf16* __restrict__ wqT, const bf16* __restrict__ wkT,
    const bf16* __restrict__ wvT, const bf16* __restrict__ wpT,
    const float* __restrict__ relb,
    float* __restrict__ out) {

    extern __shared__ __align__(16) char smem[];
    bf16* gsb  = (bf16*)(smem + OFF_GSB);    // [144][192]
    bf16* qh   = (bf16*)(smem + OFF_QH);     // [144][32]
    bf16* kh   = (bf16*)(smem + OFF_KH);     // [144][32]
    bf16* vht  = (bf16*)(smem + OFF_VHT);    // [32][160]  (V transposed)
    bf16* pbuf = (bf16*)(smem + OFF_PBUF);   // [144][160]
    bf16* ob   = (bf16*)(smem + OFF_OB);     // [144][192]

    const int b    = blockIdx.x;
    const int tid  = threadIdx.x;
    const int wave = tid >> 5;
    const int lane = tid & 31;
    const int l16  = lane & 15;
    const int hi   = lane >> 4;

    // ---- Phase A: gs -> LDS bf16, zero all K/N padding once --------------
    const float* gsp = gs + (size_t)b * (NTOK * DIM_);
    for (int i = tid; i < NTOK * DP; i += 288) {
        int r = i / DP, c = i - r * DP;
        gsb[i] = (bf16)((c < DIM_) ? gsp[r * DIM_ + c] : 0.0f);
    }
    for (int i = tid; i < NTOK * 16; i += 288)      // pbuf cols 144..159 = 0
        pbuf[(i >> 4) * NPK + 144 + (i & 15)] = (bf16)0.0f;
    for (int i = tid; i < 32 * 16; i += 288)        // vht cols 144..159 = 0
        vht[(i >> 4) * NPK + 144 + (i & 15)] = (bf16)0.0f;
    for (int i = tid; i < NTOK * 12; i += 288)      // ob cols 180..191 = 0
        ob[(i / 12) * DP + 180 + (i % 12)] = (bf16)0.0f;
    __syncthreads();

    for (int h = 0; h < HEADS_; ++h) {
        // ---- Phase B: Q/K/V projections (3 specialized passes) -----------
        proj_pass<0>(wave, l16, hi, h, gsb, wqT, bq, qh);
        proj_pass<1>(wave, l16, hi, h, gsb, wkT, bk, kh);
        proj_pass<2>(wave, l16, hi, h, gsb, wvT, bv, vht);
        __syncthreads();

        // ---- Phase C: attention; wave owns query tile row `wave` ---------
        const int m0 = wave * 16;
        const bf16* qrow = qh + (m0 + l16) * 32;
        v16bf aq = ld16(qrow + hi * 8, qrow + hi * 8 + 16);
        const float* rb = relb + h * (NTOK * NTOK);
        v8f s[9];
        #pragma unroll
        for (int nt = 0; nt < 9; ++nt) {
            int n = nt * 16 + l16;
            v8f c;
            #pragma unroll
            for (int r = 0; r < 8; ++r) c[r] = rb[(m0 + hi * 8 + r) * NTOK + n];
            const bf16* krow = kh + n * 32;
            v16bf bm = ld16(krow + hi * 8, krow + hi * 8 + 16);
            s[nt] = wmma_bf16(aq, bm, c);       // S = q k^T + rel_bias
        }
        // softmax across 144 keys (9 tiles x 16 lanes per half-wave).
        // P is stored UNnormalized (exp <= 1, bf16-safe); 1/sum is applied to
        // the O accumulator rows instead.
        float inv[8];
        #pragma unroll
        for (int r = 0; r < 8; ++r) {
            float mx = -3.0e38f;
            #pragma unroll
            for (int nt = 0; nt < 9; ++nt) mx = fmaxf(mx, s[nt][r]);
            mx = red_max16(mx);
            float sum = 0.0f;
            #pragma unroll
            for (int nt = 0; nt < 9; ++nt) {
                float e = __expf(s[nt][r] - mx);
                s[nt][r] = e;
                sum += e;
            }
            sum = red_add16(sum);
            inv[r] = 1.0f / sum;
        }
        // P -> LDS (this wave's private rows; same-wave DS ops are in-order)
        #pragma unroll
        for (int nt = 0; nt < 9; ++nt) {
            #pragma unroll
            for (int r = 0; r < 8; ++r)
                pbuf[(m0 + hi * 8 + r) * NPK + nt * 16 + l16] = (bf16)s[nt][r];
        }
        // O = P @ V  (K = 160, zero padded), then row-scale by 1/sum
        const bf16* prow = pbuf + (m0 + l16) * NPK;
        #pragma unroll
        for (int on = 0; on < 2; ++on) {
            v8f o;
            #pragma unroll
            for (int r = 0; r < 8; ++r) o[r] = 0.0f;
            const bf16* vrow = vht + (on * 16 + l16) * NPK;
            #pragma unroll
            for (int kk = 0; kk < 5; ++kk) {
                v16bf a  = ld16(prow + kk * 32 + hi * 8,  prow + kk * 32 + hi * 8 + 16);
                v16bf bm = ld16(vrow + kk * 32 + hi * 8,  vrow + kk * 32 + hi * 8 + 16);
                o = wmma_bf16(a, bm, o);
            }
            int nloc = on * 16 + l16;
            if (nloc < HD_) {
                int col = h * HD_ + nloc;
                #pragma unroll
                for (int r = 0; r < 8; ++r)
                    ob[(m0 + hi * 8 + r) * DP + col] = (bf16)(o[r] * inv[r]);
            }
        }
        __syncthreads();   // qh/kh/vht/pbuf reused by next head
    }

    // ---- Phase D: out = ob @ wp + bp  (108 tiles, 12 per wave) ----------
    float* outp = out + (size_t)b * (NTOK * DIM_);
    for (int t = wave; t < 108; t += 9) {
        int mt = t / 12, nt2 = t % 12;
        int ncol = nt2 * 16 + l16;              // 0..191
        float bval = (ncol < DIM_) ? bp[ncol] : 0.0f;
        v8f acc;
        #pragma unroll
        for (int r = 0; r < 8; ++r) acc[r] = bval;
        const bf16* arow = ob  + (mt * 16 + l16) * DP;
        const bf16* wrow = wpT + ncol * DP;
        #pragma unroll
        for (int kk = 0; kk < 6; ++kk) {
            v16bf a  = ld16(arow + kk * 32 + hi * 8,  arow + kk * 32 + hi * 8 + 16);
            v16bf bm = ld16(wrow + kk * 32 + hi * 8,  wrow + kk * 32 + hi * 8 + 16);
            acc = wmma_bf16(a, bm, acc);
        }
        if (ncol < DIM_) {
            #pragma unroll
            for (int r = 0; r < 8; ++r)
                outp[(mt * 16 + hi * 8 + r) * DIM_ + ncol] = acc[r];
        }
    }
}

// ---------------------------------------------------------------------------
extern "C" void kernel_launch(void* const* d_in, const int* in_sizes, int n_in,
                              void* d_out, int out_size, void* d_ws, size_t ws_size,
                              hipStream_t stream) {
    const float* gs = (const float*)d_in[0];
    const float* wq = (const float*)d_in[1];
    const float* bq = (const float*)d_in[2];
    const float* wk = (const float*)d_in[3];
    const float* bk = (const float*)d_in[4];
    const float* wv = (const float*)d_in[5];
    const float* bv = (const float*)d_in[6];
    const float* wp = (const float*)d_in[7];
    const float* bp = (const float*)d_in[8];
    const float* bt = (const float*)d_in[9];

    char*  ws   = (char*)d_ws;
    bf16*  wqT  = (bf16*)(ws + WS_WQT);
    bf16*  wkT  = (bf16*)(ws + WS_WKT);
    bf16*  wvT  = (bf16*)(ws + WS_WVT);
    bf16*  wpT  = (bf16*)(ws + WS_WPT);
    float* relb = (float*)(ws + WS_RELB);
    float* out  = (float*)d_out;

    // 4*192*192 + 6*144*144 = 271872 elements -> 1062 blocks of 256
    gs_attn_prep<<<1062, 256, 0, stream>>>(wq, wk, wv, wp, bt,
                                           wqT, wkT, wvT, wpT, relb);
    gs_attn_fused<<<NWIN, 288, SMEM_BYTES, stream>>>(gs, bq, bk, bv, bp,
                                                     wqT, wkT, wvT, wpT, relb, out);
}
